// MultiHeadSelfAttention_72284299591891
// MI455X (gfx1250) — compile-verified, matrix operand order checked by reading, not run on hardware
//
#include <hip/hip_runtime.h>
#include <hip/hip_bf16.h>

typedef __attribute__((ext_vector_type(16))) _Float16 v16h;
typedef __attribute__((ext_vector_type(8)))  _Float16 v8h;
typedef __attribute__((ext_vector_type(4)))  _Float16 v4h;
typedef __attribute__((ext_vector_type(2)))  _Float16 v2h;
typedef __attribute__((ext_vector_type(8)))  float    v8f;

namespace {
constexpr int Bb = 2, Ss = 2048, Ee = 2048, Aa = 2048, Hh = 16, HD = 128;
constexpr int TA = 3 * Aa; // 6144, row stride of the QKV buffer
constexpr float ROPE_THETA = 10000.f;
constexpr float QSCALE = 0.08838834764831845f; // 1/sqrt(128), folded into Q at RoPE time
}

// Load one lane's 16 f16 of a WMMA A/B fragment (ISA 7.12.2 16-bit 16x32 layout):
// lanes 0-15 hold K {0..7, 16..23}; lanes 16-31 hold K {8..15, 24..31}.
// p points at the K=0 element of this lane's row/column; rows padded so p is 16B aligned.
__device__ __forceinline__ v16h frag_ld(const _Float16* p, int laneHi) {
  const int q = laneHi ? 8 : 0;
  union { v16h v; v8h h[2]; } u;
  u.h[0] = *reinterpret_cast<const v8h*>(p + q);
  u.h[1] = *reinterpret_cast<const v8h*>(p + q + 16);
  return u.v;
}

// fp32 -> f16, 4 elements per thread (b128 load, b64 store)
__global__ __launch_bounds__(256) void cvt_f32_to_f16_x4(const float4* __restrict__ in,
                                                         v4h* __restrict__ out, int n4) {
  for (int i = blockIdx.x * blockDim.x + threadIdx.x; i < n4; i += gridDim.x * blockDim.x) {
    const float4 f = in[i];
    v4h h;
    h[0] = (_Float16)f.x; h[1] = (_Float16)f.y;
    h[2] = (_Float16)f.z; h[3] = (_Float16)f.w;
    out[i] = h;
  }
}

// In-place RoPE on Q and K sections of qkv[B][S][3][H][HD]; folds 1/sqrt(HD) into Q.
__global__ __launch_bounds__(256) void rope_inplace(_Float16* __restrict__ qkv) {
  const int n = Bb * Ss * 2 * Hh * (HD / 2);
  for (int idx = blockIdx.x * blockDim.x + threadIdx.x; idx < n; idx += gridDim.x * blockDim.x) {
    int t = idx;
    const int d2 = t & 63;  t >>= 6;   // pair index 0..63
    const int h  = t & 15;  t >>= 4;
    const int sec = t & 1;  t >>= 1;   // 0 = Q, 1 = K
    const int s = t & 2047; t >>= 11;
    const int b = t;
    v2h* p = reinterpret_cast<v2h*>(
        qkv + ((size_t)((b * Ss + s) * 3 + sec)) * Aa + h * HD + 2 * d2);
    const float inv = __powf(ROPE_THETA, -(float)(2 * d2) / (float)HD);
    const float fr = (float)s * inv;
    float sn, c;
    __sincosf(fr, &sn, &c);
    const v2h xv = *p;
    const float x1 = (float)xv[0], x2 = (float)xv[1];
    const float scl = (sec == 0) ? QSCALE : 1.0f;
    v2h ov;
    ov[0] = (_Float16)((x1 * c - x2 * sn) * scl);
    ov[1] = (_Float16)((x1 * sn + x2 * c) * scl);
    *p = ov;
  }
}

// C[M,N] = A[M,K] (f16, row-major) x B[K,N] (f16, row-major); fp32 accumulate.
// Block tile 128x128, BK=32; 8 waves in 4x2 grid, each computing 32x64 (2x4 C frags).
template <bool OUT_F16>
__global__ __launch_bounds__(256) void gemm_f16_wmma(const _Float16* __restrict__ A,
                                                     const _Float16* __restrict__ B,
                                                     void* __restrict__ Cout,
                                                     int M, int N, int K,
                                                     int lda, int ldb, int ldc) {
  constexpr int BM = 128, BN = 128, BK = 32, PAD = 8;
  __shared__ _Float16 Ash[BM][BK + PAD];   // [m][k]
  __shared__ _Float16 Bsh[BN][BK + PAD];   // [n][k]  (transposed at staging time)
  const int tid = threadIdx.x;
  const int wave = tid >> 5, lane = tid & 31;
  const int laneHi = lane >> 4, lane16 = lane & 15;
  const int wm = wave >> 1, wn = wave & 1;
  const int bm0 = blockIdx.y * BM, bn0 = blockIdx.x * BN;

  v8f acc[2][4];
  for (int i = 0; i < 2; ++i)
    for (int j = 0; j < 4; ++j)
      for (int r = 0; r < 8; ++r) acc[i][j][r] = 0.f;

  // B-staging coords: thread covers k-pair (2 rows) x 4 n
  const int bp_k = (tid >> 5) * 2;        // 0,2,..,14 (+16 on second pass)
  const int bp_n = (tid & 31) * 4;        // 0..124

  for (int k0 = 0; k0 < K; k0 += BK) {
    __syncthreads();
    // Stage A tile 128x32 (coalesced b64 row reads)
#pragma unroll
    for (int r = 0; r < 4; ++r) {
      const int m = r * 32 + (tid >> 3);
      const int kk = (tid & 7) * 4;
      *reinterpret_cast<v4h*>(&Ash[m][kk]) =
          *reinterpret_cast<const v4h*>(&A[(size_t)(bm0 + m) * lda + k0 + kk]);
    }
    // Stage B tile 32x128 transposed into [n][k]; packed pair writes (ds_store_b32)
#pragma unroll
    for (int r = 0; r < 2; ++r) {
      const int kk = r * 16 + bp_k;
      const v4h b0 = *reinterpret_cast<const v4h*>(&B[(size_t)(k0 + kk) * ldb + bn0 + bp_n]);
      const v4h b1 = *reinterpret_cast<const v4h*>(&B[(size_t)(k0 + kk + 1) * ldb + bn0 + bp_n]);
#pragma unroll
      for (int j = 0; j < 4; ++j) {
        v2h pr; pr[0] = b0[j]; pr[1] = b1[j];
        *reinterpret_cast<v2h*>(&Bsh[bp_n + j][kk]) = pr;
      }
    }
    if (k0 + BK < K) {
      __builtin_prefetch(&A[(size_t)(bm0 + (tid >> 3)) * lda + k0 + BK], 0, 1);
      __builtin_prefetch(&B[(size_t)(k0 + BK + bp_k) * ldb + bn0 + bp_n], 0, 1);
    }
    __syncthreads();

    v16h af[2], bf[4];
#pragma unroll
    for (int mf = 0; mf < 2; ++mf)
      af[mf] = frag_ld(&Ash[wm * 32 + mf * 16 + lane16][0], laneHi);
#pragma unroll
    for (int nf = 0; nf < 4; ++nf)
      bf[nf] = frag_ld(&Bsh[wn * 64 + nf * 16 + lane16][0], laneHi);
#pragma unroll
    for (int mf = 0; mf < 2; ++mf)
#pragma unroll
      for (int nf = 0; nf < 4; ++nf)
        acc[mf][nf] = __builtin_amdgcn_wmma_f32_16x16x32_f16(
            false, af[mf], false, bf[nf], (short)0, acc[mf][nf], false, false);
  }

  // C/D layout: VGPR r -> row (laneHi*8 + r), col = lane16 within each 16x16 frag.
#pragma unroll
  for (int mf = 0; mf < 2; ++mf)
#pragma unroll
    for (int nf = 0; nf < 4; ++nf)
#pragma unroll
      for (int r = 0; r < 8; ++r) {
        const int row = bm0 + wm * 32 + mf * 16 + laneHi * 8 + r;
        const int col = bn0 + wn * 64 + nf * 16 + lane16;
        const float v = acc[mf][nf][r];
        if (OUT_F16)
          reinterpret_cast<_Float16*>(Cout)[(size_t)row * ldc + col] = (_Float16)v;
        else
          reinterpret_cast<float*>(Cout)[(size_t)row * ldc + col] = v;
      }
}

// Flash attention: one block per (b, h, 128 q-rows); 8 waves x 16 q-rows each.
// Streams 32-key K/V tiles through LDS with online softmax; all matmuls via WMMA f16.
__global__ __launch_bounds__(256) void attn_fwd(const _Float16* __restrict__ qkv,
                                                _Float16* __restrict__ out) {
  __shared__ _Float16 Ksh[32][HD + 8];     // [key][d]
  __shared__ _Float16 Vsh[HD][32 + 8];     // [d][key] (transposed for B-frag reads)
  __shared__ _Float16 Psh[8][16][32 + 8];  // per-wave P: C-layout -> A-layout shuffle
  const int tid = threadIdx.x;
  const int wave = tid >> 5, lane = tid & 31;
  const int laneHi = lane >> 4, lane16 = lane & 15;
  const int qchunk = blockIdx.x & 15;      // S/128 = 16 chunks
  const int bh = blockIdx.x >> 4;
  const int h = bh & (Hh - 1), b = bh >> 4;

  const _Float16* Qb = qkv + ((size_t)(b * Ss) * 3 + 0) * Aa + h * HD;
  const _Float16* Kb = qkv + ((size_t)(b * Ss) * 3 + 1) * Aa + h * HD;
  const _Float16* Vb = qkv + ((size_t)(b * Ss) * 3 + 2) * Aa + h * HD;

  // Q fragments for this wave's 16 rows (scale already folded in by RoPE)
  const int q0 = qchunk * 128 + wave * 16;
  v16h qf[4];
  {
    const _Float16* qp = Qb + (size_t)(q0 + lane16) * TA;
#pragma unroll
    for (int c = 0; c < 4; ++c) qf[c] = frag_ld(qp + c * 32, laneHi);
  }

  v8f o[8];
  for (int nf = 0; nf < 8; ++nf)
    for (int r = 0; r < 8; ++r) o[nf][r] = 0.f;
  float mst[8], lst[8];
  for (int r = 0; r < 8; ++r) { mst[r] = -1e30f; lst[r] = 0.f; }

  // Staging coords
  const int kst_key = tid >> 3;            // K tile: 0..31
  const int kst_d0 = (tid & 7) * 16;
  const int vst_kp = (tid >> 4) * 2;       // V tile: key pair 0,2,..,30
  const int vst_d0 = (tid & 15) * 8;

  for (int kt = 0; kt < Ss; kt += 32) {
    __syncthreads();
    {
      // K tile: 32 keys x 128 d, row-major, b64 vector loads/stores
      const _Float16* kp = Kb + (size_t)(kt + kst_key) * TA + kst_d0;
#pragma unroll
      for (int j = 0; j < 4; ++j)
        *reinterpret_cast<v4h*>(&Ksh[kst_key][kst_d0 + j * 4]) =
            *reinterpret_cast<const v4h*>(kp + j * 4);
      // V tile transposed into [d][key]; packed key-pair writes (ds_store_b32)
      const _Float16* vp0 = Vb + (size_t)(kt + vst_kp) * TA + vst_d0;
      const _Float16* vp1 = vp0 + TA;
      v4h a0 = *reinterpret_cast<const v4h*>(vp0);
      v4h a1 = *reinterpret_cast<const v4h*>(vp0 + 4);
      v4h b0 = *reinterpret_cast<const v4h*>(vp1);
      v4h b1 = *reinterpret_cast<const v4h*>(vp1 + 4);
#pragma unroll
      for (int j = 0; j < 4; ++j) {
        v2h p0; p0[0] = a0[j]; p0[1] = b0[j];
        v2h p1; p1[0] = a1[j]; p1[1] = b1[j];
        *reinterpret_cast<v2h*>(&Vsh[vst_d0 + j][vst_kp]) = p0;
        *reinterpret_cast<v2h*>(&Vsh[vst_d0 + 4 + j][vst_kp]) = p1;
      }
    }
    __syncthreads();

    // S = Q(16x128) @ K^T(128x32): two 16-key subtiles x four d-chunks
    v8f sc[2];
    for (int ks = 0; ks < 2; ++ks)
      for (int r = 0; r < 8; ++r) sc[ks][r] = 0.f;
#pragma unroll
    for (int ks = 0; ks < 2; ++ks)
#pragma unroll
      for (int c = 0; c < 4; ++c) {
        const v16h bk = frag_ld(&Ksh[ks * 16 + lane16][c * 32], laneHi);
        sc[ks] = __builtin_amdgcn_wmma_f32_16x16x32_f16(
            false, qf[c], false, bk, (short)0, sc[ks], false, false);
      }

    // Online softmax. Row (laneHi*8 + r) values live across the 16 lanes of each half,
    // so xor-shuffles with mask<16 reduce within the half-wave.
    float alpha[8];
#pragma unroll
    for (int r = 0; r < 8; ++r) {
      float mv = fmaxf(sc[0][r], sc[1][r]);
#pragma unroll
      for (int x = 1; x < 16; x <<= 1) mv = fmaxf(mv, __shfl_xor(mv, x, 32));
      const float mnew = fmaxf(mst[r], mv);
      alpha[r] = __expf(mst[r] - mnew);
      float ps = 0.f;
#pragma unroll
      for (int ks = 0; ks < 2; ++ks) {
        const float p = __expf(sc[ks][r] - mnew);
        ps += p;
        Psh[wave][laneHi * 8 + r][ks * 16 + lane16] = (_Float16)p;
      }
#pragma unroll
      for (int x = 1; x < 16; x <<= 1) ps += __shfl_xor(ps, x, 32);
      lst[r] = lst[r] * alpha[r] + ps;
      mst[r] = mnew;
    }

#pragma unroll
    for (int nf = 0; nf < 8; ++nf)
#pragma unroll
      for (int r = 0; r < 8; ++r) o[nf][r] *= alpha[r];

    // O += P(16x32) @ V(32x128); P re-read from per-wave LDS in A-fragment layout
    const v16h pf = frag_ld(&Psh[wave][lane16][0], laneHi);
#pragma unroll
    for (int nf = 0; nf < 8; ++nf) {
      const v16h vf = frag_ld(&Vsh[nf * 16 + lane16][0], laneHi);
      o[nf] = __builtin_amdgcn_wmma_f32_16x16x32_f16(
          false, pf, false, vf, (short)0, o[nf], false, false);
    }
  }

  // Normalize and write out[b, s, h*HD + d] (matches transpose(0,2,1,3).reshape)
  float inv[8];
  for (int r = 0; r < 8; ++r) inv[r] = 1.f / lst[r];
  _Float16* ob = out + ((size_t)(b * Ss + q0)) * Aa + h * HD;
#pragma unroll
  for (int nf = 0; nf < 8; ++nf)
#pragma unroll
    for (int r = 0; r < 8; ++r)
      ob[(size_t)(laneHi * 8 + r) * Aa + nf * 16 + lane16] = (_Float16)(o[nf][r] * inv[r]);
}

extern "C" void kernel_launch(void* const* d_in, const int* in_sizes, int n_in,
                              void* d_out, int out_size, void* d_ws, size_t ws_size,
                              hipStream_t stream) {
  (void)in_sizes; (void)n_in; (void)out_size; (void)ws_size;
  const float* x    = (const float*)d_in[0];
  const float* Wqkv = (const float*)d_in[1];
  const float* Wout = (const float*)d_in[2];

  const size_t nx   = (size_t)Bb * Ss * Ee;  // 8.4M
  const size_t nwq  = (size_t)Ee * TA;       // 12.6M
  const size_t nwo  = (size_t)Aa * Ee;       // 4.2M
  const size_t nqkv = (size_t)Bb * Ss * TA;  // 25.2M
  const size_t nat  = (size_t)Bb * Ss * Aa;  // 8.4M

  char* ws = (char*)d_ws;
  _Float16* xh   = (_Float16*)ws; ws += nx * 2;
  _Float16* wqh  = (_Float16*)ws; ws += nwq * 2;
  _Float16* woh  = (_Float16*)ws; ws += nwo * 2;
  _Float16* qkvh = (_Float16*)ws; ws += nqkv * 2;
  _Float16* ath  = (_Float16*)ws; ws += nat * 2;

  cvt_f32_to_f16_x4<<<4096, 256, 0, stream>>>((const float4*)x, (v4h*)xh, (int)(nx / 4));
  cvt_f32_to_f16_x4<<<4096, 256, 0, stream>>>((const float4*)Wqkv, (v4h*)wqh, (int)(nwq / 4));
  cvt_f32_to_f16_x4<<<4096, 256, 0, stream>>>((const float4*)Wout, (v4h*)woh, (int)(nwo / 4));

  // QKV projection: [4096 x 2048] @ [2048 x 6144] -> f16
  gemm_f16_wmma<true><<<dim3(TA / 128, (Bb * Ss) / 128), 256, 0, stream>>>(
      xh, wqh, qkvh, Bb * Ss, TA, Ee, Ee, TA, TA);

  rope_inplace<<<4096, 256, 0, stream>>>(qkvh);

  attn_fwd<<<Bb * Hh * (Ss / 128), 256, 0, stream>>>(qkvh, ath);

  // Output projection: [4096 x 2048] @ [2048 x 2048] -> f32 directly into d_out
  gemm_f16_wmma<false><<<dim3(Ee / 128, (Bb * Ss) / 128), 256, 0, stream>>>(
      ath, woh, d_out, Bb * Ss, Ee, Aa, Aa, Ee, Ee);
}
